// Stage1_19275813224731
// MI455X (gfx1250) — matrix operand order repaired and checked by reading, not compiled, not run on hardware
//
#include <hip/hip_runtime.h>

typedef __attribute__((ext_vector_type(16))) _Float16 v16h;
typedef __attribute__((ext_vector_type(8)))  float    v8f;

#define K_OFF 27

// ---------------------------------------------------------------------------
// f32 -> f16 row conversion (pad row at index n is pre-zeroed by memset).
// ---------------------------------------------------------------------------
__global__ __launch_bounds__(256)
void cvt_f16(const float* __restrict__ in, _Float16* __restrict__ out, int count)
{
    int i = blockIdx.x * 256 + threadIdx.x;
    if (i < count) out[i] = (_Float16)in[i];
}

// ---------------------------------------------------------------------------
// Pre-swizzle weights into per-lane WMMA B fragments (f16):
//   Wsw[z][ct][kb][lane][j] = W[z][kb*32 + 16*(lane>>4) + j][ct*16 + (lane&15)]
// so each lane's 16 B values for one v_wmma_f32_16x16x32_f16 are 32 contiguous
// aligned bytes -> 2x global_load_b128 in the conv hot loop.
// ---------------------------------------------------------------------------
template<int CIN, int COUT>
__global__ __launch_bounds__(256)
void swizzle_w(const float* __restrict__ W, _Float16* __restrict__ Wsw)
{
    int i = blockIdx.x * 256 + threadIdx.x;
    if (i >= K_OFF * CIN * COUT) return;
    int j    = i & 15;
    int lane = (i >> 4) & 31;
    int rest = i >> 9;
    int kb   = rest % (CIN / 32);
    int zc   = rest / (CIN / 32);
    int ct   = zc % (COUT / 16);
    int z    = zc / (COUT / 16);
    int K    = kb * 32 + 16 * (lane >> 4) + j;
    int col  = ct * 16 + (lane & 15);
    Wsw[i] = (_Float16)W[((size_t)z * CIN + K) * COUT + col];
}

// ---------------------------------------------------------------------------
// Sparse conv, WMMA f16 path with async-LDS gather.
//   X16 : [n_in+1, CIN] f16, row n_in is all zeros (absorbs padded indices)
//   Wsw : swizzled weights (see above)
// Block = 256 threads = 8 wave32; each wave owns one 16-col N-tile (block
// covers 128 cols, grid.y = COUT/128); grid.x = row tiles; grid.z = 27.
// Gather: each lane issues global_load_async_to_lds_b128 (16B) placing its
// chunk directly in A-fragment order; s_wait_asynccnt + barrier, then the
// K-loop is 2x ds_load_b128 (A) + 2x global_load_b128 (B) + 1 WMMA per step.
// ---------------------------------------------------------------------------
template<int CIN, int COUT>
__global__ __launch_bounds__(256)
void sparse_conv_wmma(const _Float16* __restrict__ X16, int n_in,
                      const _Float16* __restrict__ Wsw,
                      const int* __restrict__ in_idx,
                      const int* __restrict__ out_idx,
                      int M, float* __restrict__ ACC, int n_out)
{
    __shared__ __align__(32) _Float16 sA[16 * CIN];   // fragment-ordered tile
    __shared__ int s_in[16];
    __shared__ int s_out[16];

    const int tid  = threadIdx.x;
    const int z    = blockIdx.z;
    const int row0 = blockIdx.x * 16;

    if (tid < 16) {
        int r  = row0 + tid;
        int gi = n_in, go = n_out;
        if (r < M) { gi = in_idx[z * M + r]; go = out_idx[z * M + r]; }
        if ((unsigned)gi > (unsigned)n_in) gi = n_in;   // -> zero pad row
        s_in[tid]  = gi;
        s_out[tid] = go;
    }
    __syncthreads();

    // Async gather: chunk idx = (row r, 8-channel group c8). Swizzled LDS slot:
    // kb = c8>>2, half = (c8>>1)&1, j0 = (c8&1)?8:0, slot lane = half*16 + r.
    constexpr int CH8 = CIN / 8;                        // 16 or 32 chunks/row
    for (int idx = tid; idx < 16 * CH8; idx += 256) {
        int r  = idx / CH8;
        int c8 = idx - r * CH8;
        int gi = s_in[r];
        const _Float16* src = X16 + (size_t)gi * CIN + c8 * 8;
        int kb   = c8 >> 2;
        int half = (c8 >> 1) & 1;
        int j0   = (c8 & 1) ? 8 : 0;
        unsigned dst = (unsigned)(uintptr_t)&sA[((kb * 32) + half * 16 + r) * 16 + j0];
        asm volatile("global_load_async_to_lds_b128 %0, %1, off"
                     :: "v"(dst), "v"((unsigned long long)(uintptr_t)src)
                     : "memory");
    }
    asm volatile("s_wait_asynccnt 0x0" ::: "memory");
    __syncthreads();

    const int wave = tid >> 5;
    const int lane = tid & 31;
    const int half = lane >> 4;
    const int lr   = lane & 15;
    const int ct   = blockIdx.y * 8 + wave;             // 16-col N-tile index
    const int col  = ct * 16 + lr;

    const _Float16* wbase =
        Wsw + ((size_t)z * (COUT / 16) + ct) * (size_t)(CIN / 32) * 512;

    v8f acc = {};
#pragma unroll
    for (int kb = 0; kb < CIN / 32; ++kb) {
        v16h a = *(const v16h*)(sA    + ((size_t)kb * 32 + lane) * 16);
        v16h b = *(const v16h*)(wbase + ((size_t)kb * 32 + lane) * 16);
        acc = __builtin_amdgcn_wmma_f32_16x16x32_f16(
                  false, a, false, b, (short)0, acc, false, false);
    }

    // D layout: VGPR v holds row (8*half + v), column = lr. Guarded scatter.
#pragma unroll
    for (int v = 0; v < 8; ++v) {
        int orow = s_out[8 * half + v];
        if (orow >= 0 && orow < n_out)
            atomicAdd(&ACC[(size_t)orow * COUT + col], acc[v]);
    }
}

// ---------------------------------------------------------------------------
// BatchNorm (training mode, population variance) + ReLU, two-pass.
// ---------------------------------------------------------------------------
template<int CH>
__global__ __launch_bounds__(256)
void bn_stats(const float* __restrict__ X, int n, float* __restrict__ stats)
{
    int g  = blockIdx.x * 256 + threadIdx.x;
    int c  = g % CH;
    int nt = (gridDim.x * 256) / CH;
    float s = 0.0f, s2 = 0.0f;
    for (int r = g / CH; r < n; r += nt) {
        float v = X[(size_t)r * CH + c];
        s += v; s2 += v * v;
    }
    atomicAdd(&stats[c], s);
    atomicAdd(&stats[CH + c], s2);
}

template<int CH>
__global__ __launch_bounds__(256)
void bn_apply_f32(float* __restrict__ X, int n, const float* __restrict__ stats,
                  const float* __restrict__ gamma, const float* __restrict__ beta)
{
    size_t i = (size_t)blockIdx.x * 256 + threadIdx.x;
    if (i >= (size_t)n * CH) return;
    int   c = (int)(i % CH);
    float m = stats[c] / (float)n;
    float var = stats[CH + c] / (float)n - m * m;
    float v = (X[i] - m) * rsqrtf(var + 1e-5f) * gamma[c] + beta[c];
    X[i] = v > 0.0f ? v : 0.0f;
}

// BN + ReLU writing the f16 mirror that feeds the next conv's async gather.
template<int CH>
__global__ __launch_bounds__(256)
void bn_apply_f16(const float* __restrict__ X, int n, const float* __restrict__ stats,
                  const float* __restrict__ gamma, const float* __restrict__ beta,
                  _Float16* __restrict__ OUT)
{
    size_t i = (size_t)blockIdx.x * 256 + threadIdx.x;
    if (i >= (size_t)n * CH) return;
    int   c = (int)(i % CH);
    float m = stats[c] / (float)n;
    float var = stats[CH + c] / (float)n - m * m;
    float v = (X[i] - m) * rsqrtf(var + 1e-5f) * gamma[c] + beta[c];
    OUT[i] = (_Float16)(v > 0.0f ? v : 0.0f);
}

// ---------------------------------------------------------------------------
// segment_max over post-ReLU (>=0) features: int-bitwise atomicMax is order-
// preserving for non-negative floats; OUT pre-zeroed (valid identity here).
// ---------------------------------------------------------------------------
__global__ __launch_bounds__(256)
void seg_max128(const float* __restrict__ X, const int* __restrict__ parent,
                int n, float* __restrict__ OUT)
{
    size_t i = (size_t)blockIdx.x * 256 + threadIdx.x;
    if (i >= (size_t)n * 128) return;
    int r = (int)(i / 128), c = (int)(i % 128);
    int p = parent[r];
    atomicMax((int*)&OUT[(size_t)p * 128 + c], __float_as_int(X[i]));
}

// ---------------------------------------------------------------------------
extern "C" void kernel_launch(void* const* d_in, const int* in_sizes, int n_in,
                              void* d_out, int out_size, void* d_ws, size_t ws_size,
                              hipStream_t stream)
{
    const float* feats  = (const float*)d_in[0];
    const float* W1     = (const float*)d_in[1];
    const float* g1     = (const float*)d_in[2];
    const float* b1     = (const float*)d_in[3];
    const float* W2     = (const float*)d_in[4];
    const float* g2     = (const float*)d_in[5];
    const float* b2     = (const float*)d_in[6];
    const float* W3     = (const float*)d_in[7];
    const float* g3     = (const float*)d_in[8];
    const float* b3     = (const float*)d_in[9];
    const float* W4     = (const float*)d_in[10];
    const int*   in1    = (const int*)d_in[11];
    const int*   out1   = (const int*)d_in[12];
    const int*   in2    = (const int*)d_in[13];
    const int*   out2   = (const int*)d_in[14];
    const int*   parent = (const int*)d_in[15];
    // d_in[16] (n_pool) lives on device; recover it from out_size instead.

    const int N      = in_sizes[0] / 128;
    const int M1     = in_sizes[11] / K_OFF;
    const int M2     = in_sizes[13] / K_OFF;
    const int n_pool = out_size / 256;
    float* out = (float*)d_out;

    // Carve 32B-aligned scratch.
    char* p = (char*)d_ws;
    auto carve = [&](size_t bytes) {
        char* r = p;
        p += (bytes + 31) & ~(size_t)31;
        return r;
    };
    _Float16* feats16 = (_Float16*)carve((size_t)(N + 1) * 128 * 2);
    float*    acc1    = (float*)   carve((size_t)N * 128 * 4);
    float*    accP    = (float*)   carve((size_t)n_pool * 128 * 4);
    _Float16* accP16  = (_Float16*)carve((size_t)(n_pool + 1) * 128 * 2);
    float*    acc2    = (float*)   carve((size_t)n_pool * 256 * 4);
    _Float16* x216    = (_Float16*)carve((size_t)(n_pool + 1) * 256 * 2);
    float*    acc3    = (float*)   carve((size_t)n_pool * 256 * 4);
    _Float16* x316    = (_Float16*)carve((size_t)(n_pool + 1) * 256 * 2);
    _Float16* W1sw    = (_Float16*)carve((size_t)K_OFF * 128 * 128 * 2);
    _Float16* W2sw    = (_Float16*)carve((size_t)K_OFF * 128 * 256 * 2);
    _Float16* W3sw    = (_Float16*)carve((size_t)K_OFF * 256 * 256 * 2);
    _Float16* W4sw    = (_Float16*)carve((size_t)K_OFF * 256 * 256 * 2);
    float*    stats   = (float*)   carve(512 * 4);

    // Zero accumulators, stats, output, and the f16 pad rows.
    hipMemsetAsync(acc1,  0, (size_t)N * 128 * 4, stream);
    hipMemsetAsync(accP,  0, (size_t)n_pool * 128 * 4, stream);
    hipMemsetAsync(acc2,  0, (size_t)n_pool * 256 * 4, stream);
    hipMemsetAsync(acc3,  0, (size_t)n_pool * 256 * 4, stream);
    hipMemsetAsync(stats, 0, 512 * 4, stream);
    hipMemsetAsync(out,   0, (size_t)out_size * 4, stream);
    hipMemsetAsync(feats16 + (size_t)N * 128,      0, 128 * 2, stream);
    hipMemsetAsync(accP16  + (size_t)n_pool * 128, 0, 128 * 2, stream);
    hipMemsetAsync(x216    + (size_t)n_pool * 256, 0, 256 * 2, stream);
    hipMemsetAsync(x316    + (size_t)n_pool * 256, 0, 256 * 2, stream);

    const dim3 blk(256);
    auto blocks = [](size_t elems) { return (int)((elems + 255) / 256); };

    // One-time per-call prep: f16 feature mirror + swizzled f16 weights.
    cvt_f16<<<blocks((size_t)N * 128), blk, 0, stream>>>(feats, feats16, N * 128);
    swizzle_w<128, 128><<<blocks((size_t)K_OFF * 128 * 128), blk, 0, stream>>>(W1, W1sw);
    swizzle_w<128, 256><<<blocks((size_t)K_OFF * 128 * 256), blk, 0, stream>>>(W2, W2sw);
    swizzle_w<256, 256><<<blocks((size_t)K_OFF * 256 * 256), blk, 0, stream>>>(W3, W3sw);
    swizzle_w<256, 256><<<blocks((size_t)K_OFF * 256 * 256), blk, 0, stream>>>(W4, W4sw);

    // Layer 1: conv(128->128) + BN/ReLU + maxpool (+ f16 mirror for conv2).
    sparse_conv_wmma<128, 128>
        <<<dim3((M1 + 15) / 16, 1, K_OFF), blk, 0, stream>>>(
            feats16, N, W1sw, in1, out1, M1, acc1, N);
    bn_stats<128><<<256, blk, 0, stream>>>(acc1, N, stats);
    bn_apply_f32<128><<<blocks((size_t)N * 128), blk, 0, stream>>>(
        acc1, N, stats, g1, b1);
    seg_max128<<<blocks((size_t)N * 128), blk, 0, stream>>>(acc1, parent, N, accP);
    cvt_f16<<<blocks((size_t)n_pool * 128), blk, 0, stream>>>(
        accP, accP16, n_pool * 128);

    // Layer 2: conv(128->256) + BN/ReLU -> f16 mirror.
    hipMemsetAsync(stats, 0, 512 * 4, stream);
    sparse_conv_wmma<128, 256>
        <<<dim3((M2 + 15) / 16, 2, K_OFF), blk, 0, stream>>>(
            accP16, n_pool, W2sw, in2, out2, M2, acc2, n_pool);
    bn_stats<256><<<256, blk, 0, stream>>>(acc2, n_pool, stats);
    bn_apply_f16<256><<<blocks((size_t)n_pool * 256), blk, 0, stream>>>(
        acc2, n_pool, stats, g2, b2, x216);

    // Layer 3: conv(256->256) + BN/ReLU -> f16 mirror.
    hipMemsetAsync(stats, 0, 512 * 4, stream);
    sparse_conv_wmma<256, 256>
        <<<dim3((M2 + 15) / 16, 2, K_OFF), blk, 0, stream>>>(
            x216, n_pool, W3sw, in2, out2, M2, acc3, n_pool);
    bn_stats<256><<<256, blk, 0, stream>>>(acc3, n_pool, stats);
    bn_apply_f16<256><<<blocks((size_t)n_pool * 256), blk, 0, stream>>>(
        acc3, n_pool, stats, g3, b3, x316);

    // Layer 4: conv(256->256) straight into d_out (pre-zeroed; pads guarded).
    sparse_conv_wmma<256, 256>
        <<<dim3((M2 + 15) / 16, 2, K_OFF), blk, 0, stream>>>(
            x316, n_pool, W4sw, in2, out2, M2, out, n_pool);
}
